// MultiHeadAttention_56599079026961
// MI455X (gfx1250) — matrix operand order repaired and checked by reading, not compiled, not run on hardware
//
#include <hip/hip_runtime.h>
#include <hip/hip_bf16.h>

// ---------------------------------------------------------------------------
// MHA for MI455X (gfx1250, wave32, WMMA).
//   B=2, S=4096, D=512, H=8, Dh=64.
// Round 4: attention wave now owns 32 queries (2 WMMA column tiles), so every
// K/V fragment load feeds 2 WMMAs -> 16 WMMA per 16 b128 loads per chunk
// (2x arithmetic intensity, half the K/V re-read traffic).
// ---------------------------------------------------------------------------

typedef __bf16 bf16_t;
typedef __attribute__((ext_vector_type(16))) __bf16 v16bf;
typedef __attribute__((ext_vector_type(8)))  __bf16 v8bf;
typedef __attribute__((ext_vector_type(4)))  __bf16 v4bf;
typedef __attribute__((ext_vector_type(2)))  __bf16 v2bf;
typedef __attribute__((ext_vector_type(8)))  float  v8f;
typedef __attribute__((ext_vector_type(8)))  unsigned int v8u;

#define D_MODEL 512
#define SEQ     4096
#define HEADS   8
#define DH      64
#define BATCH   2

__device__ __forceinline__ v8f wmma_bf16(v16bf a, v16bf b, v8f c) {
  // D = A(16x32) * B(32x16) + C, fp32 accumulate
  return __builtin_amdgcn_wmma_f32_16x16x32_bf16(
      /*neg_a=*/false, a, /*neg_b=*/false, b,
      /*c_mod=*/(short)0, c, /*reuse_a=*/false, /*reuse_b=*/false);
}

// ---- A fragment (16x32, bf16) from row-major [row][k], row stride ld.
// Lane l: row = l&15; kb = (l&16)?16:0; holds k = kb..kb+15.
__device__ __forceinline__ v16bf load_a_frag(const bf16_t* base, int ld, int lane) {
  int r  = lane & 15;
  int kb = (lane & 16) ? 16 : 0;
  const bf16_t* p = base + (size_t)r * ld + kb;
  v8bf lo = *(const v8bf*)(p);
  v8bf hi = *(const v8bf*)(p + 8);
  v16bf a;
#pragma unroll
  for (int i = 0; i < 8; ++i) { a[i] = lo[i]; a[8 + i] = hi[i]; }
  return a;
}

__device__ __forceinline__ v16bf load_a_frag(const float* base, int ld, int lane) {
  int r  = lane & 15;
  int kb = (lane & 16) ? 16 : 0;
  const float* p = base + (size_t)r * ld + kb;
  v16bf a;
#pragma unroll
  for (int i = 0; i < 8; ++i) { a[i] = (bf16_t)p[i]; a[8 + i] = (bf16_t)p[8 + i]; }
  return a;
}

// ---- B fragment (32x16, bf16) where B(k,n) = src[n*ld + k] ("n-major").
// Lane l: col n = l&15; kb = (l&16)?16:0; vector element j is k = kb+j.
__device__ __forceinline__ v16bf load_b_frag(const bf16_t* base, int ld, int lane) {
  int n  = lane & 15;
  int kb = (lane & 16) ? 16 : 0;
  const bf16_t* p = base + (size_t)n * ld + kb;
  return *(const v16bf*)p;   // 32B contiguous
}

__device__ __forceinline__ v16bf load_b_frag(const float* base, int ld, int lane) {
  int n  = lane & 15;
  int kb = (lane & 16) ? 16 : 0;
  const float* p = base + (size_t)n * ld + kb;
  v16bf b;
#pragma unroll
  for (int i = 0; i < 16; ++i) b[i] = (bf16_t)p[i];
  return b;
}

__device__ __forceinline__ unsigned int pk_bf16(float a, float b) {
  v2bf t; t[0] = (bf16_t)a; t[1] = (bf16_t)b;
  return __builtin_bit_cast(unsigned int, t);
}

// ---------------------------------------------------------------------------
// Pre-conversion kernels: fp32 -> bf16, 4 elements/thread, vectorized.
// ---------------------------------------------------------------------------
__global__ __launch_bounds__(256)
void mha_cvt_qkv(const float* __restrict__ a, const float* __restrict__ b,
                 const float* __restrict__ c, bf16_t* __restrict__ da,
                 bf16_t* __restrict__ db, bf16_t* __restrict__ dc, int n) {
  const float* s = (blockIdx.z == 0) ? a : (blockIdx.z == 1) ? b : c;
  bf16_t*      d = (blockIdx.z == 0) ? da : (blockIdx.z == 1) ? db : dc;
  int i = (blockIdx.x * 256 + threadIdx.x) * 4;
  if (i < n) {
    float4 f = *(const float4*)(s + i);
    v4bf o;
    o[0] = (bf16_t)f.x; o[1] = (bf16_t)f.y; o[2] = (bf16_t)f.z; o[3] = (bf16_t)f.w;
    *(v4bf*)(d + i) = o;
  }
}

__global__ __launch_bounds__(256)
void mha_cvt_w(const float* __restrict__ a, const float* __restrict__ b,
               const float* __restrict__ c, const float* __restrict__ e,
               bf16_t* __restrict__ da, bf16_t* __restrict__ db,
               bf16_t* __restrict__ dc, bf16_t* __restrict__ de, int n) {
  const float* s = (blockIdx.z == 0) ? a : (blockIdx.z == 1) ? b
                 : (blockIdx.z == 2) ? c : e;
  bf16_t*      d = (blockIdx.z == 0) ? da : (blockIdx.z == 1) ? db
                 : (blockIdx.z == 2) ? dc : de;
  int i = (blockIdx.x * 256 + threadIdx.x) * 4;
  if (i < n) {
    float4 f = *(const float4*)(s + i);
    v4bf o;
    o[0] = (bf16_t)f.x; o[1] = (bf16_t)f.y; o[2] = (bf16_t)f.z; o[3] = (bf16_t)f.w;
    *(v4bf*)(d + i) = o;
  }
}

// ---------------------------------------------------------------------------
// Kernel 1: QKV projection. Y = X @ W^T + b  (M=8192, N=512, K=512)
//   z==0: Q -> Qh [B][H][S][DH] bf16, pre-scaled by 1/sqrt(DH)
//   z==1: K -> Kh [B][H][S][DH] bf16
//   z==2: V -> VhT [B][H][DH][S] bf16 (transposed; used by O^T = V^T P^T)
// ---------------------------------------------------------------------------
template <typename TX, typename TW>
__global__ __launch_bounds__(256)
void mha_qkv_proj(const TX* __restrict__ q, const TX* __restrict__ k,
                  const TX* __restrict__ v,
                  const TW* __restrict__ Wq, const TW* __restrict__ Wk,
                  const TW* __restrict__ Wv,
                  const float* __restrict__ bq, const float* __restrict__ bk,
                  const float* __restrict__ bv,
                  bf16_t* __restrict__ Qh, bf16_t* __restrict__ Kh,
                  bf16_t* __restrict__ VhT) {
  const int lane = threadIdx.x & 31;
  const int wave = threadIdx.x >> 5;
  const int mode = blockIdx.z;               // 0=Q 1=K 2=V
  const int m0   = blockIdx.x * 128 + wave * 16;
  const int n0   = blockIdx.y * 64;

  const TX* x    = (mode == 0) ? q  : (mode == 1) ? k  : v;
  const TW* W    = (mode == 0) ? Wq : (mode == 1) ? Wk : Wv;
  const float* bias = (mode == 0) ? bq : (mode == 1) ? bk : bv;

  v8f acc[4] = {v8f{}, v8f{}, v8f{}, v8f{}};

#pragma unroll 2
  for (int kk = 0; kk < D_MODEL; kk += 32) {
    v16bf a = load_a_frag(x + (size_t)m0 * D_MODEL + kk, D_MODEL, lane);
#pragma unroll
    for (int t = 0; t < 4; ++t) {
      v16bf b = load_b_frag(W + (size_t)(n0 + t * 16) * D_MODEL + kk, D_MODEL, lane);
      acc[t] = wmma_bf16(a, b, acc[t]);
    }
  }

  const int col  = lane & 15;
  const int half = lane >> 4;
#pragma unroll
  for (int t = 0; t < 4; ++t) {
    int   n  = n0 + t * 16 + col;
    float bn = bias[n];
    int   h  = n >> 6;        // head
    int   dh = n & 63;        // feature in head
#pragma unroll
    for (int i = 0; i < 8; ++i) {
      int m = m0 + i + 8 * half;
      int b = m >> 12;        // batch (S=4096)
      int s = m & (SEQ - 1);
      float val = acc[t][i] + bn;
      if (mode == 0) {
        val *= 0.125f;        // 1/sqrt(64) folded into Q
        Qh[(((size_t)b * HEADS + h) * SEQ + s) * DH + dh] = (bf16_t)val;
      } else if (mode == 1) {
        Kh[(((size_t)b * HEADS + h) * SEQ + s) * DH + dh] = (bf16_t)val;
      } else {
        VhT[(((size_t)b * HEADS + h) * DH + dh) * SEQ + s] = (bf16_t)val;
      }
    }
  }
}

// ---------------------------------------------------------------------------
// Kernel 2: transposed flash attention per (b,h), 2 query tiles per wave.
// Wave owns 32 queries (two WMMA column tiles). Keys stream in chunks of 32.
//   S^T tiles: A = K rows (keys x feat) shared by both query sets -> 8 WMMA
//   softmax: lane-local reduce + 1 shfl_xor(16) per query set
//   P^T B-fragments: packed bf16 half-swap via 4 shfl_xor(16) per set
//   O^T += V^T P^T: each V fragment feeds both sets               -> 8 WMMA
// Grid (S/256, B*H), block 256. No LDS tiles.
// ---------------------------------------------------------------------------
__global__ __launch_bounds__(256)
void mha_attn(const bf16_t* __restrict__ Qh, const bf16_t* __restrict__ Kh,
              const bf16_t* __restrict__ VhT, bf16_t* __restrict__ ctx) {
  const int lane = threadIdx.x & 31;
  const int wave = threadIdx.x >> 5;
  const int bh   = blockIdx.y;                 // b*H + h
  const int q0   = blockIdx.x * 256 + wave * 32;
  const int half = lane >> 4;

  const bf16_t* Qb = Qh  + (size_t)bh * SEQ * DH;
  const bf16_t* Kb = Kh  + (size_t)bh * SEQ * DH;
  const bf16_t* Vb = VhT + (size_t)bh * DH * SEQ;

  // Q^T as B operands (64 features x 16 queries), per query set, loaded once
  v16bf bq[2][2];
#pragma unroll
  for (int qs = 0; qs < 2; ++qs) {
    bq[qs][0] = load_b_frag(Qb + (size_t)(q0 + qs * 16) * DH,      DH, lane);
    bq[qs][1] = load_b_frag(Qb + (size_t)(q0 + qs * 16) * DH + 32, DH, lane);
  }

  float mq[2] = {-1e30f, -1e30f};   // running max per lane's query, per set
  float lq[2] = {0.0f, 0.0f};       // running sum
  v8f acc[2][4] = {{v8f{}, v8f{}, v8f{}, v8f{}}, {v8f{}, v8f{}, v8f{}, v8f{}}};

  for (int kc = 0; kc < SEQ; kc += 32) {
    // ---- S^T tiles: s[qs][0] = keys kc..+15, s[qs][1] = keys kc+16..+31
    v8f s[2][2] = {{v8f{}, v8f{}}, {v8f{}, v8f{}}};
    {
      v16bf ak = load_a_frag(Kb + (size_t)kc * DH, DH, lane);
      s[0][0] = wmma_bf16(ak, bq[0][0], s[0][0]);
      s[1][0] = wmma_bf16(ak, bq[1][0], s[1][0]);
      ak = load_a_frag(Kb + (size_t)kc * DH + 32, DH, lane);
      s[0][0] = wmma_bf16(ak, bq[0][1], s[0][0]);
      s[1][0] = wmma_bf16(ak, bq[1][1], s[1][0]);
      ak = load_a_frag(Kb + (size_t)(kc + 16) * DH, DH, lane);
      s[0][1] = wmma_bf16(ak, bq[0][0], s[0][1]);
      s[1][1] = wmma_bf16(ak, bq[1][0], s[1][1]);
      ak = load_a_frag(Kb + (size_t)(kc + 16) * DH + 32, DH, lane);
      s[0][1] = wmma_bf16(ak, bq[0][1], s[0][1]);
      s[1][1] = wmma_bf16(ak, bq[1][1], s[1][1]);
    }
    // lane's rows: s[qs][0][i] -> key kc+i+8*half ; s[qs][1][i] -> key kc+16+i+8*half

    v16bf pb[2];
#pragma unroll
    for (int qs = 0; qs < 2; ++qs) {
      // ---- chunk max (lane-local over 16, + one cross-half shfl)
      float mx = s[qs][0][0];
#pragma unroll
      for (int i = 1; i < 8; ++i) mx = fmaxf(mx, s[qs][0][i]);
#pragma unroll
      for (int i = 0; i < 8; ++i) mx = fmaxf(mx, s[qs][1][i]);
      mx = fmaxf(mx, __shfl_xor(mx, 16, 32));

      float mnew = fmaxf(mq[qs], mx);
      float corr = __expf(mq[qs] - mnew);
      mq[qs] = mnew;

      // ---- exponentiate + chunk sum
      float p0[8], p1[8];
      float rs = 0.0f;
#pragma unroll
      for (int i = 0; i < 8; ++i) {
        p0[i] = __expf(s[qs][0][i] - mnew);
        p1[i] = __expf(s[qs][1][i] - mnew);
        rs += p0[i] + p1[i];
      }
      rs += __shfl_xor(rs, 16, 32);
      lq[qs] = lq[qs] * corr + rs;

      // ---- rescale O^T accumulator (corr uniform per lane = per query)
#pragma unroll
      for (int t = 0; t < 4; ++t)
#pragma unroll
        for (int i = 0; i < 8; ++i) acc[qs][t][i] *= corr;

      // ---- build P^T B-fragment (32 keys x 16 queries) via packed half-swap.
      unsigned int own0[4], own1[4], recv[4];
#pragma unroll
      for (int g = 0; g < 4; ++g) {
        own0[g] = pk_bf16(p0[2 * g], p0[2 * g + 1]);   // keys  8*half+2g..+1
        own1[g] = pk_bf16(p1[2 * g], p1[2 * g + 1]);   // keys 16+8*half+2g..+1
      }
#pragma unroll
      for (int g = 0; g < 4; ++g) {
        unsigned int send = half ? own0[g] : own1[g];
        recv[g] = (unsigned int)__shfl_xor((int)send, 16, 32);
      }
      v8u pu;
#pragma unroll
      for (int g = 0; g < 4; ++g) {
        pu[g]     = half ? recv[g] : own0[g];  // keys kb+0..7
        pu[4 + g] = half ? own1[g] : recv[g];  // keys kb+8..15
      }
      pb[qs] = __builtin_bit_cast(v16bf, pu);
    }

    // ---- O^T += V^T(feat x keys) @ P^T(keys x queries); V shared by both sets
#pragma unroll
    for (int t = 0; t < 4; ++t) {
      v16bf av = load_a_frag(Vb + (size_t)(t * 16) * SEQ + kc, SEQ, lane);
      acc[0][t] = wmma_bf16(av, pb[0], acc[0][t]);
      acc[1][t] = wmma_bf16(av, pb[1], acc[1][t]);
    }
  }

  // ---- finalize: O^T C-layout: lane col = query, rows = features i+8*half.
  // Per lane: token fixed, 8 consecutive features per slab -> 16B stores.
  const int b = bh >> 3;
  const int h = bh & 7;
#pragma unroll
  for (int qs = 0; qs < 2; ++qs) {
    const int s = q0 + qs * 16 + (lane & 15);
    float inv = 1.0f / lq[qs];
    bf16_t* crow = ctx + ((size_t)b * SEQ + s) * D_MODEL + h * DH + 8 * half;
#pragma unroll
    for (int t = 0; t < 4; ++t) {
      v8bf o;
#pragma unroll
      for (int i = 0; i < 8; ++i) o[i] = (bf16_t)(acc[qs][t][i] * inv);
      *(v8bf*)(crow + t * 16) = o;
    }
  }
}

// ---------------------------------------------------------------------------
// Kernel 3: output projection. out = ctx @ Wo^T + bo (fp32 out).
// ---------------------------------------------------------------------------
template <typename TW>
__global__ __launch_bounds__(256)
void mha_out_proj(const bf16_t* __restrict__ ctx, const TW* __restrict__ Wo,
                  const float* __restrict__ bo, float* __restrict__ out) {
  const int lane = threadIdx.x & 31;
  const int wave = threadIdx.x >> 5;
  const int m0   = blockIdx.x * 128 + wave * 16;
  const int n0   = blockIdx.y * 64;

  v8f acc[4] = {v8f{}, v8f{}, v8f{}, v8f{}};
#pragma unroll 2
  for (int kk = 0; kk < D_MODEL; kk += 32) {
    v16bf a = load_a_frag(ctx + (size_t)m0 * D_MODEL + kk, D_MODEL, lane);
#pragma unroll
    for (int t = 0; t < 4; ++t) {
      v16bf b = load_b_frag(Wo + (size_t)(n0 + t * 16) * D_MODEL + kk, D_MODEL, lane);
      acc[t] = wmma_bf16(a, b, acc[t]);
    }
  }

  const int col  = lane & 15;
  const int half = lane >> 4;
#pragma unroll
  for (int t = 0; t < 4; ++t) {
    int   n  = n0 + t * 16 + col;
    float bn = bo[n];
#pragma unroll
    for (int i = 0; i < 8; ++i) {
      int m = m0 + i + 8 * half;
      out[(size_t)m * D_MODEL + n] = acc[t][i] + bn;
    }
  }
}

// ---------------------------------------------------------------------------
extern "C" void kernel_launch(void* const* d_in, const int* in_sizes, int n_in,
                              void* d_out, int out_size, void* d_ws, size_t ws_size,
                              hipStream_t stream) {
  (void)in_sizes; (void)n_in; (void)out_size;
  const float* q  = (const float*)d_in[0];
  const float* k  = (const float*)d_in[1];
  const float* v  = (const float*)d_in[2];
  const float* Wq = (const float*)d_in[3];
  const float* bq = (const float*)d_in[4];
  const float* Wk = (const float*)d_in[5];
  const float* bk = (const float*)d_in[6];
  const float* Wv = (const float*)d_in[7];
  const float* bv = (const float*)d_in[8];
  const float* Wo = (const float*)d_in[9];
  const float* bo = (const float*)d_in[10];
  float* out = (float*)d_out;

  char* ws = (char*)d_ws;
  const size_t REG = (size_t)BATCH * SEQ * D_MODEL * sizeof(bf16_t); // 8 MB
  const size_t WSZ = (size_t)D_MODEL * D_MODEL * sizeof(bf16_t);     // 512 KB
  bf16_t* Qh  = (bf16_t*)(ws + 0 * REG);
  bf16_t* Kh  = (bf16_t*)(ws + 1 * REG);
  bf16_t* VhT = (bf16_t*)(ws + 2 * REG);
  bf16_t* ctx = (bf16_t*)(ws + 3 * REG);
  // fast-path bf16 copies
  bf16_t* Xq  = (bf16_t*)(ws + 4 * REG);
  bf16_t* Xk  = (bf16_t*)(ws + 5 * REG);
  bf16_t* Xv  = (bf16_t*)(ws + 6 * REG);
  bf16_t* Wqb = (bf16_t*)(ws + 7 * REG + 0 * WSZ);
  bf16_t* Wkb = (bf16_t*)(ws + 7 * REG + 1 * WSZ);
  bf16_t* Wvb = (bf16_t*)(ws + 7 * REG + 2 * WSZ);
  bf16_t* Wob = (bf16_t*)(ws + 7 * REG + 3 * WSZ);
  const size_t NEED_FAST = 7 * REG + 4 * WSZ;   // ~58 MB

  dim3 blk(256);
  const int NX = BATCH * SEQ * D_MODEL;   // 4,194,304
  const int NW = D_MODEL * D_MODEL;       // 262,144

  if (ws_size >= NEED_FAST) {
    mha_cvt_qkv<<<dim3(NX / 1024, 1, 3), blk, 0, stream>>>(q, k, v, Xq, Xk, Xv, NX);
    mha_cvt_w<<<dim3(NW / 1024, 1, 4), blk, 0, stream>>>(Wq, Wk, Wv, Wo,
                                                         Wqb, Wkb, Wvb, Wob, NW);
    mha_qkv_proj<bf16_t, bf16_t><<<dim3(64, 8, 3), blk, 0, stream>>>(
        Xq, Xk, Xv, Wqb, Wkb, Wvb, bq, bk, bv, Qh, Kh, VhT);
    mha_attn<<<dim3(16, 16), blk, 0, stream>>>(Qh, Kh, VhT, ctx);
    mha_out_proj<bf16_t><<<dim3(64, 8), blk, 0, stream>>>(ctx, Wob, bo, out);
  } else {
    mha_qkv_proj<float, float><<<dim3(64, 8, 3), blk, 0, stream>>>(
        q, k, v, Wq, Wk, Wv, bq, bk, bv, Qh, Kh, VhT);
    mha_attn<<<dim3(16, 16), blk, 0, stream>>>(Qh, Kh, VhT, ctx);
    mha_out_proj<float><<<dim3(64, 8), blk, 0, stream>>>(ctx, Wo, bo, out);
  }
}
